// CXNGeneralLayer_19696720019799
// MI455X (gfx1250) — compile-verified, matrix operand order checked by reading, not run on hardware
//
#include <hip/hip_runtime.h>

typedef __attribute__((ext_vector_type(16))) _Float16 v16h;
typedef __attribute__((ext_vector_type(8)))  float    v8f;

#define N_ROWS 8192   // N_I == N_J == N_K
#define C_OUT  32
#define C_IN   32
#define KTOT   32768  // 4 * 8192

// ---------------------------------------------------------------------------
// Prep: Y_m = x_m @ W_m + b_m for the 4 branches, written as f16 directly in
// WMMA B-fragment order: Yfrag[kblk][nhalf][lane][slot] (16 halfs / lane).
// k = concatenated K index (0..32767); element (k, n) goes to:
//   lane = (n & 15) + 16 * ((k >> 3) & 1)
//   slot = 8 * (k >> 4 & 1...) -> within a 32-K block: s = 8*(kin/16) + (kin & 7)
// ---------------------------------------------------------------------------
__global__ __launch_bounds__(256) void prep_linears(
    const float* __restrict__ xi,  const float* __restrict__ xj1,
    const float* __restrict__ xk,
    const float* __restrict__ Wi,  const float* __restrict__ Wj1,
    const float* __restrict__ Wj2, const float* __restrict__ Wk,
    const float* __restrict__ bi,  const float* __restrict__ bj1,
    const float* __restrict__ bj2, const float* __restrict__ bk,
    _Float16* __restrict__ Yfrag)
{
    int tid = blockIdx.x * 256 + threadIdx.x;     // 32768*32 threads
    int k   = tid >> 5;                           // concatenated row 0..32767
    int n   = tid & 31;                           // output channel
    int branch = k >> 13;
    int row    = k & (N_ROWS - 1);

    const float *x, *W, *b;
    if      (branch == 0) { x = xi;  W = Wi;  b = bi;  }
    else if (branch == 1) { x = xj1; W = Wj1; b = bj1; }
    else if (branch == 2) { x = xj1; W = Wj2; b = bj2; }   // faithful: xj1 w/ W_j2
    else                  { x = xk;  W = Wk;  b = bk;  }

    float acc = b[n];
    const float* xr = x + (size_t)row * C_IN;
    #pragma unroll
    for (int c = 0; c < C_IN; ++c)
        acc += xr[c] * W[c * C_OUT + n];

    int kblk = k >> 5;
    int kin  = k & 31;
    int lane = (n & 15) + (((k >> 3) & 1) << 4);
    int slot = (((kin >> 4) & 1) << 3) | (kin & 7);
    size_t idx = ((((size_t)kblk << 1) + (n >> 4)) * 32 + lane) * 16 + slot;
    Yfrag[idx] = (_Float16)acc;
}

// ---------------------------------------------------------------------------
// Main: C[8192,32] = [G0|G1|G2|G3] @ Y, ReLU, f32 out.
// Block = 8 waves, one 16-row tile. Wave w: matrix m = w>>1, col-half = w&1,
// 256 steps of v_wmma_f32_16x16x32_f16 over its 8192-wide K partition.
// ---------------------------------------------------------------------------
__global__ __launch_bounds__(256) void merge_gemm_relu(
    const float* __restrict__ G0, const float* __restrict__ G1,
    const float* __restrict__ G2, const float* __restrict__ G3,
    const _Float16* __restrict__ Yfrag,
    float* __restrict__ out)
{
    __shared__ float red[8 * 32 * 8];   // 8 KB

    int lane = threadIdx.x & 31;
    int w    = threadIdx.x >> 5;
    int m    = w >> 1;                  // which G matrix (K partition)
    int nh   = w & 1;                   // column half (0: N 0-15, 1: N 16-31)

    const float* G = (m == 0) ? G0 : (m == 1) ? G1 : (m == 2) ? G2 : G3;

    int rowTile = blockIdx.x;           // 0..511
    int row = rowTile * 16 + (lane & 15);
    int khi = (lane >> 4) << 3;         // lanes 16-31 hold K+8 / K+24 groups

    const float*    arow = G + (size_t)row * N_ROWS + khi;
    const _Float16* bptr = Yfrag
        + ((((size_t)m * 512) + nh) * 32 + lane) * 16;  // kblk = m*256

    v8f acc = {};

    #pragma unroll 2
    for (int kb = 0; kb < 256; ++kb) {
        // A tile: 16 rows x 32 K of G, f32 -> f16 (layout per ISA 16-bit A)
        float4 a0 = *(const float4*)(arow);
        float4 a1 = *(const float4*)(arow + 4);
        float4 a2 = *(const float4*)(arow + 16);
        float4 a3 = *(const float4*)(arow + 20);
        __builtin_prefetch((const void*)(arow + 256), 0, 3);

        v16h A;
        A[0]  = (_Float16)a0.x; A[1]  = (_Float16)a0.y;
        A[2]  = (_Float16)a0.z; A[3]  = (_Float16)a0.w;
        A[4]  = (_Float16)a1.x; A[5]  = (_Float16)a1.y;
        A[6]  = (_Float16)a1.z; A[7]  = (_Float16)a1.w;
        A[8]  = (_Float16)a2.x; A[9]  = (_Float16)a2.y;
        A[10] = (_Float16)a2.z; A[11] = (_Float16)a2.w;
        A[12] = (_Float16)a3.x; A[13] = (_Float16)a3.y;
        A[14] = (_Float16)a3.z; A[15] = (_Float16)a3.w;

        v16h B = *(const v16h*)(bptr);  // pre-swizzled, coalesced b128 x2

        acc = __builtin_amdgcn_wmma_f32_16x16x32_f16(
            /*neg_a=*/false, A, /*neg_b=*/false, B,
            /*c_mod=*/(short)0, acc, /*reuse_a=*/false, /*reuse_b=*/false);

        arow += 32;
        bptr += 1024;   // 32 K * 2 halves * 32 lanes * 16 halfs / (2*32*16)=... next kblk
    }

    // 4-way K-partition reduction across waves via LDS, then ReLU + store.
    #pragma unroll
    for (int r = 0; r < 8; ++r)
        red[(w * 32 + lane) * 8 + r] = acc[r];
    __syncthreads();

    if (w < 2) {   // w==nh here: wave 0 -> cols 0-15, wave 1 -> cols 16-31
        int col   = (lane & 15) + (w << 4);
        int rbase = rowTile * 16 + ((lane >> 4) << 3);
        #pragma unroll
        for (int r = 0; r < 8; ++r) {
            float v = red[((w + 0) * 32 + lane) * 8 + r]
                    + red[((w + 2) * 32 + lane) * 8 + r]
                    + red[((w + 4) * 32 + lane) * 8 + r]
                    + red[((w + 6) * 32 + lane) * 8 + r];
            out[(size_t)(rbase + r) * C_OUT + col] = v > 0.0f ? v : 0.0f;
        }
    }
}

// ---------------------------------------------------------------------------
extern "C" void kernel_launch(void* const* d_in, const int* in_sizes, int n_in,
                              void* d_out, int out_size, void* d_ws, size_t ws_size,
                              hipStream_t stream)
{
    const float* xi     = (const float*)d_in[0];
    const float* xj1    = (const float*)d_in[1];
    // d_in[2] (xj2) intentionally unused — faithful to reference.
    const float* xk     = (const float*)d_in[3];
    const float* Gi2j   = (const float*)d_in[4];
    const float* Adj2j  = (const float*)d_in[5];
    const float* coAdj  = (const float*)d_in[6];
    const float* Gk2j   = (const float*)d_in[7];
    const float* Wi     = (const float*)d_in[8];
    const float* Wj1    = (const float*)d_in[9];
    const float* Wj2    = (const float*)d_in[10];
    const float* Wk     = (const float*)d_in[11];
    const float* bi     = (const float*)d_in[12];
    const float* bj1    = (const float*)d_in[13];
    const float* bj2    = (const float*)d_in[14];
    const float* bk     = (const float*)d_in[15];

    _Float16* Yfrag = (_Float16*)d_ws;   // 32768*32 halfs = 2 MB

    prep_linears<<<(KTOT * C_OUT) / 256, 256, 0, stream>>>(
        xi, xj1, xk, Wi, Wj1, Wj2, Wk, bi, bj1, bj2, bk, Yfrag);

    merge_gemm_relu<<<N_ROWS / 16, 256, 0, stream>>>(
        Gi2j, Adj2j, coAdj, Gk2j, Yfrag, (float*)d_out);
}
